// Seq2SeqModel_28750511079943
// MI455X (gfx1250) — matrix operand order repaired and checked by reading, not compiled, not run on hardware
//
#include <hip/hip_runtime.h>
#include <hip/hip_bf16.h>

#define B_ 256
#define T_ 128
#define F_ 32
#define H_ 1024
#define G_ 4096   // 4*H

typedef __bf16 bf16_t;
typedef __attribute__((ext_vector_type(16))) __bf16 v16bf;
typedef __attribute__((ext_vector_type(8)))  float  v8f;

union FragAB { uint4 u[2]; v16bf v; };
union FragC  { v8f v; float f[8]; };

__device__ __forceinline__ float sig_(float x)  { return 1.0f / (1.0f + __expf(-x)); }
__device__ __forceinline__ float tanh_(float x) { return 1.0f - 2.0f / (__expf(2.0f * x) + 1.0f); }

// CDNA5 async global->LDS copy (ASYNCcnt-tracked, bypasses VGPRs).
// lds_off = wave-relative LDS byte address (low 32 bits of generic shared ptr).
__device__ __forceinline__ void async_copy_b128(unsigned lds_off, const void* gaddr) {
    asm volatile("global_load_async_to_lds_b128 %0, %1, off"
                 :: "v"(lds_off), "v"(gaddr)
                 : "memory");
}

// ---------------------------------------------------------------------------
// gates[256,4096] = X[256,KX] @ Wih[4096,KX]^T + Hprev[256,1024] @ Whh[4096,1024]^T + bias
// Tile: WG = 128(M) x 64(N), 128 threads = 4 waves in 2x2, wave tile = 64(M) x 32(N).
// Double-buffered LDS fed by global_load_async_to_lds_b128; single fused chunk
// stream over the [X | Hprev] K dimension so the pipeline never drains.
// KX is a template constant so the pipelined loop has compile-time bounds.
// ---------------------------------------------------------------------------
template<int KX>
__global__ __launch_bounds__(128)
void gates_gemm_kernel(const bf16_t* __restrict__ X, int ldx,
                       const bf16_t* __restrict__ Wih,
                       const bf16_t* __restrict__ Hprev,
                       const bf16_t* __restrict__ Whh,
                       const float*  __restrict__ bias,
                       float*        __restrict__ gates)
{
    __shared__ bf16_t ldsA[2][128 * 40];   // 128 rows x 32 K, pitch 40 (bank pad)
    __shared__ bf16_t ldsB[2][64 * 40];    // 64  rows x 32 K

    constexpr int NX  = KX >> 5;           // K-chunks from the X stream
    constexpr int NCH = NX + (H_ >> 5);    // + K-chunks from the Hprev stream

    const int tid  = threadIdx.x;
    const int wave = tid >> 5;
    const int lane = tid & 31;
    const int l16  = lane & 15;
    const int hi   = lane >> 4;

    const int blockM = blockIdx.y * 128;
    const int blockN = blockIdx.x * 64;
    const int waveM  = (wave >> 1) * 64;
    const int waveN  = (wave & 1) * 32;

    // wave-relative LDS byte offsets for the async-copy destinations
    const unsigned ldsA0 = (unsigned)(size_t)&ldsA[0][0];
    const unsigned ldsA1 = (unsigned)(size_t)&ldsA[1][0];
    const unsigned ldsB0 = (unsigned)(size_t)&ldsB[0][0];
    const unsigned ldsB1 = (unsigned)(size_t)&ldsB[1][0];
    const unsigned laOff = (unsigned)tid * 80u;
    const int      rB    = tid >> 1;
    const int      ks    = (tid & 1) * 16;
    const unsigned lbOff = (unsigned)rB * 80u + (unsigned)ks * 2u;

    // per-thread global stream pointers, computed once (no muls in the loop)
    const bf16_t* paX = X     + (blockM + tid) * ldx;
    const bf16_t* paH = Hprev + (blockM + tid) * H_;
    const bf16_t* pbX = Wih   + (blockN + rB) * KX + ks;
    const bf16_t* pbH = Whh   + (blockN + rB) * H_ + ks;

    FragC acc[4][2];
#pragma unroll
    for (int i = 0; i < 4; ++i)
#pragma unroll
        for (int j = 0; j < 2; ++j)
#pragma unroll
            for (int r = 0; r < 8; ++r) acc[i][j].f[r] = 0.0f;

    // issue 6 async b128 copies for chunk ic into buffer (ic & 1)
    auto prefetch = [&](int ic) {
        const int buf = ic & 1;
        const bf16_t* srcA;
        const bf16_t* srcB;
        if (ic < NX) { srcA = paX + ic * 32;        srcB = pbX + ic * 32; }
        else         { srcA = paH + (ic - NX) * 32; srcB = pbH + (ic - NX) * 32; }
        // A tile: one row (32 bf16 = 64B) per thread
        const unsigned la = (buf ? ldsA1 : ldsA0) + laOff;
        async_copy_b128(la,       srcA);
        async_copy_b128(la + 16u, srcA + 8);
        async_copy_b128(la + 32u, srcA + 16);
        async_copy_b128(la + 48u, srcA + 24);
        // B tile: 64 rows, 2 threads per row (32B each)
        const unsigned lb = (buf ? ldsB1 : ldsB0) + lbOff;
        async_copy_b128(lb,       srcB);
        async_copy_b128(lb + 16u, srcB + 8);
    };

    prefetch(0);

    for (int ic = 0; ic < NCH; ++ic) {
        const int buf = ic & 1;
        if (ic + 1 < NCH) {
            prefetch(ic + 1);
            // chunk ic's 6 copies retired; chunk ic+1's 6 may remain in flight
            asm volatile("s_wait_asynccnt 0x6" ::: "memory");
        } else {
            asm volatile("s_wait_asynccnt 0x0" ::: "memory");
        }
        __syncthreads();

        const bf16_t* bA = buf ? ldsA[1] : ldsA[0];
        const bf16_t* bB = buf ? ldsB[1] : ldsB[0];

        FragAB bfr[2];
#pragma unroll
        for (int j = 0; j < 2; ++j) {
            // B 32x16: lane n = column; lanes 0-15 K 0..7 & 16..23, lanes 16-31 K 8..15 & 24..31
            const bf16_t* p = &bB[(waveN + j * 16 + l16) * 40 + hi * 8];
            bfr[j].u[0] = *(const uint4*)(p);
            bfr[j].u[1] = *(const uint4*)(p + 16);
        }
#pragma unroll
        for (int i = 0; i < 4; ++i) {
            // A 16x32: lane m = row; lanes 0-15 K 0..7 & 16..23, lanes 16-31 K 8..15 & 24..31
            FragAB afr;
            const bf16_t* p = &bA[(waveM + i * 16 + l16) * 40 + hi * 8];
            afr.u[0] = *(const uint4*)(p);
            afr.u[1] = *(const uint4*)(p + 16);
#pragma unroll
            for (int j = 0; j < 2; ++j) {
                acc[i][j].v = __builtin_amdgcn_wmma_f32_16x16x32_bf16(
                    false, afr.v, false, bfr[j].v, (short)0, acc[i][j].v, false, false);
            }
        }
        __syncthreads();   // all waves done reading buf before it is refilled
    }

    // epilogue: add bias, store fp32 gates
#pragma unroll
    for (int j = 0; j < 2; ++j) {
        const int col = blockN + waveN + j * 16 + l16;
        const float bv = bias[col];
#pragma unroll
        for (int i = 0; i < 4; ++i) {
            const int mbase = blockM + waveM + i * 16 + hi * 8;
#pragma unroll
            for (int r = 0; r < 8; ++r) {
                gates[(mbase + r) * G_ + col] = acc[i][j].f[r] + bv;
            }
        }
    }
}

// ---------------------------------------------------------------------------
// Elementwise LSTM cell: c_new = f*c + i*tanh(g); h_new = o*tanh(c_new) (bf16)
// ---------------------------------------------------------------------------
__global__ __launch_bounds__(256)
void lstm_cell_kernel(const float* __restrict__ gates,
                      float*       __restrict__ c,
                      bf16_t*      __restrict__ h_bf)
{
    const int idx = blockIdx.x * 256 + threadIdx.x;   // [0, B_*H_)
    const int b  = idx >> 10;
    const int hh = idx & (H_ - 1);
    const float* g = gates + b * G_;
    const float i_ = sig_(g[hh]);
    const float f_ = sig_(g[H_ + hh]);
    const float gg = tanh_(g[2 * H_ + hh]);
    const float o_ = sig_(g[3 * H_ + hh]);
    const float cn = f_ * c[idx] + i_ * gg;
    c[idx] = cn;
    h_bf[idx] = (bf16_t)(o_ * tanh_(cn));
}

// ---------------------------------------------------------------------------
// Decoder output projection: out[256,32] = h1 @ linW[32,1024]^T + lin_b
// One wave per 16x16 tile; writes fp32 into d_out[:, t, :] and bf16 next input
// ---------------------------------------------------------------------------
__global__ __launch_bounds__(32)
void linear_out_kernel(const bf16_t* __restrict__ h1,
                       const bf16_t* __restrict__ Wl,   // [F_, H_]
                       const float*  __restrict__ bl,
                       float*        __restrict__ out,  // [B_, T_, F_]
                       bf16_t*       __restrict__ xdec, // [B_, F_]
                       int t)
{
    const int lane = threadIdx.x;
    const int l16 = lane & 15, hi = lane >> 4;
    const int mt = blockIdx.x >> 1;   // 0..15
    const int nt = blockIdx.x & 1;    // 0..1

    FragC acc;
#pragma unroll
    for (int r = 0; r < 8; ++r) acc.f[r] = 0.0f;

    for (int kc = 0; kc < H_; kc += 32) {
        FragAB a, b;
        const bf16_t* pa = h1 + (mt * 16 + l16) * H_ + kc + hi * 8;
        a.u[0] = *(const uint4*)(pa);
        a.u[1] = *(const uint4*)(pa + 16);
        const bf16_t* pb = Wl + (nt * 16 + l16) * H_ + kc + hi * 8;
        b.u[0] = *(const uint4*)(pb);
        b.u[1] = *(const uint4*)(pb + 16);
        acc.v = __builtin_amdgcn_wmma_f32_16x16x32_bf16(
            false, a.v, false, b.v, (short)0, acc.v, false, false);
    }

    const int col = nt * 16 + l16;
    const float bv = bl[col];
    const int mbase = mt * 16 + hi * 8;
#pragma unroll
    for (int r = 0; r < 8; ++r) {
        const float v = acc.f[r] + bv;
        const int m = mbase + r;
        out[m * (T_ * F_) + t * F_ + col] = v;
        xdec[m * F_ + col] = (bf16_t)v;
    }
}

// ---------------------------------------------------------------------------
// Utility kernels
// ---------------------------------------------------------------------------
__global__ void cvt_bf16_kernel(const float* __restrict__ src, bf16_t* __restrict__ dst, int n) {
    const int i = blockIdx.x * 256 + threadIdx.x;
    if (i < n) dst[i] = (bf16_t)src[i];
}
__global__ void zero_f32_kernel(float* p, int n) {
    const int i = blockIdx.x * 256 + threadIdx.x;
    if (i < n) p[i] = 0.0f;
}
__global__ void zero_bf16_kernel(bf16_t* p, int n) {
    const int i = blockIdx.x * 256 + threadIdx.x;
    if (i < n) p[i] = (bf16_t)0.0f;
}

// ---------------------------------------------------------------------------
extern "C" void kernel_launch(void* const* d_in, const int* in_sizes, int n_in,
                              void* d_out, int out_size, void* d_ws, size_t ws_size,
                              hipStream_t stream)
{
    const float* in_seq   = (const float*)d_in[0];
    const float* enc_Wih0 = (const float*)d_in[1];
    const float* enc_Whh0 = (const float*)d_in[2];
    const float* enc_b0   = (const float*)d_in[3];
    const float* enc_Wih1 = (const float*)d_in[4];
    const float* enc_Whh1 = (const float*)d_in[5];
    const float* enc_b1   = (const float*)d_in[6];
    const float* dec_Wih0 = (const float*)d_in[7];
    const float* dec_Whh0 = (const float*)d_in[8];
    const float* dec_b0   = (const float*)d_in[9];
    const float* dec_Wih1 = (const float*)d_in[10];
    const float* dec_Whh1 = (const float*)d_in[11];
    const float* dec_b1   = (const float*)d_in[12];
    const float* lin_W    = (const float*)d_in[13];
    const float* lin_b    = (const float*)d_in[14];
    float* out = (float*)d_out;

    // workspace bump allocator (256B aligned)
    char* ws = (char*)d_ws;
    size_t off = 0;
    auto alloc = [&](size_t bytes) -> char* {
        char* p = ws + off;
        off += (bytes + 255) & ~(size_t)255;
        return p;
    };

    bf16_t* inBf   = (bf16_t*)alloc((size_t)B_ * T_ * F_ * 2);
    bf16_t* eWih0  = (bf16_t*)alloc((size_t)G_ * F_ * 2);
    bf16_t* eWhh0  = (bf16_t*)alloc((size_t)G_ * H_ * 2);
    bf16_t* eWih1  = (bf16_t*)alloc((size_t)G_ * H_ * 2);
    bf16_t* eWhh1  = (bf16_t*)alloc((size_t)G_ * H_ * 2);
    bf16_t* dWih0  = (bf16_t*)alloc((size_t)G_ * F_ * 2);
    bf16_t* dWhh0  = (bf16_t*)alloc((size_t)G_ * H_ * 2);
    bf16_t* dWih1  = (bf16_t*)alloc((size_t)G_ * H_ * 2);
    bf16_t* dWhh1  = (bf16_t*)alloc((size_t)G_ * H_ * 2);
    bf16_t* linWbf = (bf16_t*)alloc((size_t)F_ * H_ * 2);
    bf16_t* h0bf   = (bf16_t*)alloc((size_t)B_ * H_ * 2);
    bf16_t* h1bf   = (bf16_t*)alloc((size_t)B_ * H_ * 2);
    bf16_t* xdec   = (bf16_t*)alloc((size_t)B_ * F_ * 2);
    float*  c0     = (float*)alloc((size_t)B_ * H_ * 4);
    float*  c1     = (float*)alloc((size_t)B_ * H_ * 4);
    float*  gates  = (float*)alloc((size_t)B_ * G_ * 4);
    (void)ws_size; (void)in_sizes; (void)n_in; (void)out_size;

    auto cvt = [&](const float* s, bf16_t* d, int n) {
        cvt_bf16_kernel<<<(n + 255) / 256, 256, 0, stream>>>(s, d, n);
    };

    // one-time conversions
    cvt(in_seq,   inBf,   B_ * T_ * F_);
    cvt(enc_Wih0, eWih0,  G_ * F_);
    cvt(enc_Whh0, eWhh0,  G_ * H_);
    cvt(enc_Wih1, eWih1,  G_ * H_);
    cvt(enc_Whh1, eWhh1,  G_ * H_);
    cvt(dec_Wih0, dWih0,  G_ * F_);
    cvt(dec_Whh0, dWhh0,  G_ * H_);
    cvt(dec_Wih1, dWih1,  G_ * H_);
    cvt(dec_Whh1, dWhh1,  G_ * H_);
    cvt(lin_W,    linWbf, F_ * H_);

    // init state
    zero_f32_kernel <<<(B_ * H_ + 255) / 256, 256, 0, stream>>>(c0, B_ * H_);
    zero_f32_kernel <<<(B_ * H_ + 255) / 256, 256, 0, stream>>>(c1, B_ * H_);
    zero_bf16_kernel<<<(B_ * H_ + 255) / 256, 256, 0, stream>>>(h0bf, B_ * H_);
    zero_bf16_kernel<<<(B_ * H_ + 255) / 256, 256, 0, stream>>>(h1bf, B_ * H_);

    const dim3 gemm_grid(G_ / 64, B_ / 128);   // 64 x 2 = 128 workgroups
    const int  cell_grid = (B_ * H_) / 256;

    // ---------------- encoder ----------------
    for (int t = 0; t < T_; ++t) {
        gates_gemm_kernel<F_><<<gemm_grid, 128, 0, stream>>>(
            inBf + (size_t)t * F_, T_ * F_, eWih0, h0bf, eWhh0, enc_b0, gates);
        lstm_cell_kernel<<<cell_grid, 256, 0, stream>>>(gates, c0, h0bf);

        gates_gemm_kernel<H_><<<gemm_grid, 128, 0, stream>>>(
            h0bf, H_, eWih1, h1bf, eWhh1, enc_b1, gates);
        lstm_cell_kernel<<<cell_grid, 256, 0, stream>>>(gates, c1, h1bf);
    }

    // ---------------- decoder ----------------
    for (int t = 0; t < T_; ++t) {
        const bf16_t* x = (t == 0) ? (inBf + (size_t)(T_ - 1) * F_) : xdec;
        const int ldx   = (t == 0) ? (T_ * F_) : F_;

        gates_gemm_kernel<F_><<<gemm_grid, 128, 0, stream>>>(
            x, ldx, dWih0, h0bf, dWhh0, dec_b0, gates);
        lstm_cell_kernel<<<cell_grid, 256, 0, stream>>>(gates, c0, h0bf);

        gates_gemm_kernel<H_><<<gemm_grid, 128, 0, stream>>>(
            h0bf, H_, dWih1, h1bf, dWhh1, dec_b1, gates);
        lstm_cell_kernel<<<cell_grid, 256, 0, stream>>>(gates, c1, h1bf);

        linear_out_kernel<<<32, 32, 0, stream>>>(h1bf, linWbf, lin_b, out, xdec, t);
    }
}
